// GraphChebyConv_69475390980636
// MI455X (gfx1250) — compile-verified, matrix operand order checked by reading, not run on hardware
//
#include <hip/hip_runtime.h>

// ---------------------------------------------------------------------------
// Chebyshev graph convolution for MI455X (gfx1250, wave32, WMMA).
//
// Strategy:
//   1. Convert L (8192x8192 fp32, 256MB) to f16 once -> 128MB, resident in
//      the 192MB L2 for all 24 recurrence passes.
//   2. 24 sequential GEMMs Y = L @ X (8192x8192 @ 8192x32) using
//      v_wmma_f32_16x16x32_f16, fp32 accumulation. K split 8 ways into
//      deterministic partial buffers (no fp atomics).
//   3. Epilogue per step: reduce partials in fixed order, apply
//      X_k = 2Y - X_{k-2} (X_1 = Y), emit f16-transposed operand for the
//      next step + fp32 copy for the final projection.
//   4. Final projection (32768x200)@(200x16) + bias, weights staged in LDS.
// ---------------------------------------------------------------------------

typedef __attribute__((ext_vector_type(16))) _Float16 v16h;
typedef __attribute__((ext_vector_type(4)))  _Float16 v4h;
typedef __attribute__((ext_vector_type(8)))  float    v8f;

#define GN   8192          // graph nodes
#define GC   32            // B * F_IN = 4*8 columns
#define NC   (GN * GC)     // 262144 elements per X buffer
#define KORD 25            // Chebyshev order
#define KSPLIT 8           // K-dimension split for parallelism
#define KCHUNK (GN / KSPLIT)   // 1024

// --- 1. L fp32 -> f16 (64M elements, 4 per thread, vectorized) -------------
__global__ void __launch_bounds__(256)
cheb_convert_l(const float* __restrict__ L, _Float16* __restrict__ Lh) {
    size_t i = ((size_t)blockIdx.x * 256 + threadIdx.x) * 4;
    float4 v = *(const float4*)(L + i);
    v4h h;
    h[0] = (_Float16)v.x; h[1] = (_Float16)v.y;
    h[2] = (_Float16)v.z; h[3] = (_Float16)v.w;
    *(v4h*)(Lh + i) = h;
}

// --- 2. Init: X0^T(f16) = x (elementwise, since x is (b,fin,n) flat),
//        Xs[0] fp32 in (n, c) layout for the projection ---------------------
__global__ void __launch_bounds__(256)
cheb_init_x(const float* __restrict__ x, _Float16* __restrict__ XT0,
            float* __restrict__ Xs0) {
    int e = blockIdx.x * 256 + threadIdx.x;   // e = c*GN + n
    int c = e >> 13;
    int n = e & (GN - 1);
    float v = x[e];
    XT0[e] = (_Float16)v;
    Xs0[(size_t)n * GC + c] = v;
}

// --- 3. WMMA GEMM: Ypart[kc] += L[32 rows] @ X[32 cols] over a 1024 K-chunk.
//        One wave = 2x2 grid of 16x16 f32 accum tiles, K loop of 32 WMMAs x4.
__global__ void __launch_bounds__(256)
cheb_gemm_step(const _Float16* __restrict__ Lh, const _Float16* __restrict__ XT,
               float* __restrict__ Ypart) {
    const int lane  = threadIdx.x & 31;
    const int wave  = threadIdx.x >> 5;
    const int gw    = blockIdx.x * 8 + wave;     // 0..2047
    const int mblk  = gw & 255;                  // 256 row-blocks of 32
    const int kc    = gw >> 8;                   // 0..7 K-chunk
    const int row16 = lane & 15;
    const int hl    = lane >> 4;                 // lane half: K-group selector
    const int m0    = mblk * 32;
    const int k0    = kc * KCHUNK;

    // A: rows of L (row-major, stride GN). Two 16B loads per lane per tile:
    //   elems 0..7  = K k+hl*8 .. +7 ; elems 8..15 = K k+16+hl*8 .. +7
    const _Float16* a0p = Lh + (size_t)(m0 + row16) * GN + k0 + hl * 8;
    const _Float16* a1p = a0p + (size_t)16 * GN;
    // B: columns of X = rows of XT (32 x 8192 row-major), same striping.
    const _Float16* b0p = XT + (size_t)row16 * GN + k0 + hl * 8;
    const _Float16* b1p = b0p + (size_t)16 * GN;

    v8f acc00 = {}, acc01 = {}, acc10 = {}, acc11 = {};

    for (int kk = 0; kk < KCHUNK; kk += 32) {
        v16h a0, a1, b0, b1;
        ((uint4*)&a0)[0] = *(const uint4*)(a0p + kk);
        ((uint4*)&a0)[1] = *(const uint4*)(a0p + kk + 16);
        ((uint4*)&a1)[0] = *(const uint4*)(a1p + kk);
        ((uint4*)&a1)[1] = *(const uint4*)(a1p + kk + 16);
        ((uint4*)&b0)[0] = *(const uint4*)(b0p + kk);
        ((uint4*)&b0)[1] = *(const uint4*)(b0p + kk + 16);
        ((uint4*)&b1)[0] = *(const uint4*)(b1p + kk);
        ((uint4*)&b1)[1] = *(const uint4*)(b1p + kk + 16);

        acc00 = __builtin_amdgcn_wmma_f32_16x16x32_f16(false, a0, false, b0,
                                                       (short)0, acc00, false, false);
        acc01 = __builtin_amdgcn_wmma_f32_16x16x32_f16(false, a0, false, b1,
                                                       (short)0, acc01, false, false);
        acc10 = __builtin_amdgcn_wmma_f32_16x16x32_f16(false, a1, false, b0,
                                                       (short)0, acc10, false, false);
        acc11 = __builtin_amdgcn_wmma_f32_16x16x32_f16(false, a1, false, b1,
                                                       (short)0, acc11, false, false);
    }

    // C/D layout: vgpr v, lane -> (M = v + hl*8, N = lane&15). Plain stores to
    // this K-chunk's private partial slot => deterministic reduction later.
    float* yp = Ypart + (size_t)kc * NC;
    const int cA = row16, cB = row16 + 16;
#pragma unroll
    for (int v = 0; v < 8; ++v) {
        int mA = m0 + v + hl * 8;
        int mB = mA + 16;
        yp[(size_t)mA * GC + cA] = acc00[v];
        yp[(size_t)mA * GC + cB] = acc01[v];
        yp[(size_t)mB * GC + cA] = acc10[v];
        yp[(size_t)mB * GC + cB] = acc11[v];
    }
}

// --- 4. Epilogue: fixed-order reduce 8 partials, Chebyshev recurrence,
//        write next f16 transposed operand + fp32 projection copy -----------
__global__ void __launch_bounds__(256)
cheb_epilogue(const float* __restrict__ Ypart, const float* __restrict__ Xm2,
              _Float16* __restrict__ XTout, float* __restrict__ Xsout,
              int isFirst) {
    int e = blockIdx.x * 256 + threadIdx.x;   // e = n*GC + c
    int n = e >> 5;
    int c = e & 31;
    float y = 0.0f;
#pragma unroll
    for (int j = 0; j < KSPLIT; ++j) y += Ypart[(size_t)j * NC + e];
    float xv = isFirst ? y : (2.0f * y - Xm2[e]);
    Xsout[e] = xv;
    XTout[(size_t)c * GN + n] = (_Float16)xv;
}

// --- 5. Projection: out[b,f,n] = bias[f] + sum_{fin,k} Xs[k][n, b*8+fin] *
//        W[fin*25+k, f], weights staged in LDS -------------------------------
__global__ void __launch_bounds__(256)
cheb_project(const float* __restrict__ Xs, const float* __restrict__ W,
             const float* __restrict__ bias, float* __restrict__ out) {
    __shared__ float w[200 * 16];
    for (int i = threadIdx.x; i < 200 * 16; i += 256) w[i] = W[i];
    __syncthreads();

    int gid = blockIdx.x * 256 + threadIdx.x;  // 0..32767 = b*GN + n
    int b = gid >> 13;
    int n = gid & (GN - 1);

    float acc[16];
#pragma unroll
    for (int f = 0; f < 16; ++f) acc[f] = bias[f];

    for (int k = 0; k < KORD; ++k) {
        const float* xp = Xs + (size_t)k * NC + (size_t)n * GC + b * 8;
#pragma unroll
        for (int fin = 0; fin < 8; ++fin) {
            float xv = xp[fin];
            const float* wr = &w[(fin * KORD + k) * 16];
#pragma unroll
            for (int f = 0; f < 16; ++f) acc[f] += xv * wr[f];
        }
    }
#pragma unroll
    for (int f = 0; f < 16; ++f)
        out[((size_t)(b * 16 + f)) * GN + n] = acc[f];
}

// ---------------------------------------------------------------------------
extern "C" void kernel_launch(void* const* d_in, const int* in_sizes, int n_in,
                              void* d_out, int out_size, void* d_ws, size_t ws_size,
                              hipStream_t stream) {
    const float* x    = (const float*)d_in[0];   // (4, 8, 8192)
    const float* L    = (const float*)d_in[1];   // (8192, 8192)
    const float* W    = (const float*)d_in[2];   // (200, 16)
    const float* bias = (const float*)d_in[3];   // (16,)
    float* out = (float*)d_out;                  // (4, 16, 8192)

    char* ws = (char*)d_ws;
    // ws layout (bytes):
    //   [0,            128 MB)  Lh      : f16 L
    //   [128 MB,      +8 MB  )  Ypart   : 8 fp32 partial (8192x32) buffers
    //   [.. ,         +1.5 MB)  XT[3]   : rotating f16 transposed X (32x8192)
    //   [.. ,        +25 MB  )  Xs[25]  : fp32 X_k in (n, c) layout
    _Float16* Lh    = (_Float16*)ws;
    float*    Ypart = (float*)(ws + (size_t)134217728);
    _Float16* XT    = (_Float16*)(ws + (size_t)134217728 + 8388608);
    float*    Xs    = (float*)(ws + (size_t)134217728 + 8388608 + 3 * (size_t)NC * 2);

    // 1. L -> f16 (fits in 192MB L2 for all subsequent passes)
    cheb_convert_l<<<65536, 256, 0, stream>>>(L, Lh);

    // 2. X0: f16 transpose buffer + fp32 projection slot 0
    cheb_init_x<<<NC / 256, 256, 0, stream>>>(x, XT, Xs);

    // 3. Chebyshev recurrence: 24 WMMA GEMM passes over L2-resident Lh
    for (int k = 1; k < KORD; ++k) {
        const _Float16* xin  = XT + (size_t)((k - 1) % 3) * NC;
        _Float16*       xout = XT + (size_t)(k % 3) * NC;
        cheb_gemm_step<<<256, 256, 0, stream>>>(Lh, xin, Ypart);
        cheb_epilogue<<<NC / 256, 256, 0, stream>>>(
            Ypart, Xs + (size_t)(k >= 2 ? k - 2 : 0) * NC,
            xout, Xs + (size_t)k * NC, (k == 1) ? 1 : 0);
    }

    // 4. Output projection + bias
    cheb_project<<<128, 256, 0, stream>>>(Xs, W, bias, out);
}
